// CausalFieldAttention_57629871178547
// MI455X (gfx1250) — matrix operand order repaired and checked by reading, not compiled, not run on hardware
//
#include <hip/hip_runtime.h>

// ---------------------------------------------------------------------------
// CausalFieldAttention for MI455X (gfx1250, wave32, WMMA bf16)
//   B=4, N=4096, D=1024, H=16, hd=64, G=512
//   q-projection in the reference is dead code -> skipped.
//   conv done directly (256 causal taps), not via FFT.
//   GEMMs: LDS double-buffered, fed by gfx1250 async global->LDS loads
//   (ASYNCcnt / s_wait_asynccnt) when the toolchain exposes the builtins.
// Workspace layout (needs 52 MB):
//   [ 0MB) x_bf16      32MB
//   [32MB) k_w bf16     2MB
//   [34MB) v_w bf16     2MB
//   [36MB) out_w bf16   2MB
//   [40MB) field f32    8MB   (B,H,G,hd)
//   [48MB) conv bf16    4MB   (B,G,D)
// ---------------------------------------------------------------------------

typedef __bf16 bf16;
typedef __attribute__((ext_vector_type(16))) __bf16 v16bf;
typedef __attribute__((ext_vector_type(8)))  __bf16 v8bf;
typedef __attribute__((ext_vector_type(8)))  float  v8f;
typedef __attribute__((ext_vector_type(4)))  int    v4i;

typedef __attribute__((address_space(1))) v4i* as1_v4i_ptr;
typedef __attribute__((address_space(3))) v4i* as3_v4i_ptr;

#define D_EMB  1024
#define NHEADS 16
#define HD     64
#define GFIELD 512
#define NSEQ   4096
#define BATCH  4
#define MTOT   (BATCH * NSEQ)   // 16384
#define KSTEPS (D_EMB / 32)     // 32

#if defined(__has_builtin)
#if __has_builtin(__builtin_amdgcn_global_load_async_to_lds_b128) && \
    __has_builtin(__builtin_amdgcn_s_wait_asynccnt)
#define CFA_ASYNC 1
#endif
#endif
#ifndef CFA_ASYNC
#define CFA_ASYNC 0
#endif

__device__ __forceinline__ v8f wmma_bf16(v16bf a, v16bf b, v8f c) {
  return __builtin_amdgcn_wmma_f32_16x16x32_bf16(false, a, false, b,
                                                 (short)0, c, false, false);
}

// Move 16 bytes global -> LDS. Async (ASYNCcnt) when available, else a
// register relay (global_load_b128 + ds_store_b128) ordered by __syncthreads.
__device__ __forceinline__ void copy16_g2l(const bf16* g, bf16* l) {
#if CFA_ASYNC
  v4i* gv = const_cast<v4i*>(reinterpret_cast<const v4i*>(g));
  v4i* lv = reinterpret_cast<v4i*>(l);
  __builtin_amdgcn_global_load_async_to_lds_b128(
      (as1_v4i_ptr)gv, (as3_v4i_ptr)lv, 0, 0);
#else
  *reinterpret_cast<v8bf*>(l) = *reinterpret_cast<const v8bf*>(g);
#endif
}

__device__ __forceinline__ void wait_g2l() {
#if CFA_ASYNC
  __builtin_amdgcn_s_wait_asynccnt(0);
#endif
}

// A fragment from an LDS tile (64 rows x 32 k, row-major, ld=32).
// ISA 16-bit A 16x32 layout: lane<16 row=lane, K[kb..kb+8) & K[kb+16..kb+24),
// kb = 8*(lane>=16).
__device__ __forceinline__ v16bf lds_frag_a(const bf16* sA, int rs, int lane) {
  const int kb = (lane >> 4) << 3;
  const bf16* p = sA + (rs * 16 + (lane & 15)) * 32 + kb;
  union { v16bf v; v8bf h[2]; } u;
  u.h[0] = *reinterpret_cast<const v8bf*>(p);
  u.h[1] = *reinterpret_cast<const v8bf*>(p + 16);
  return u.v;
}

// B fragment from an LDS weight tile (64 cols x 32 k, row-major W[n,k], ld=32).
// Lane holds column t16+(lane&15), K[kb..kb+8) & K[kb+16..kb+24), kb = 8*(lane>=16).
__device__ __forceinline__ v16bf lds_frag_b(const bf16* sW, int t16, int lane) {
  const int kb = (lane >> 4) << 3;
  const bf16* p = sW + (t16 + (lane & 15)) * 32 + kb;
  union { v16bf v; v8bf h[2]; } u;
  u.h[0] = *reinterpret_cast<const v8bf*>(p);
  u.h[1] = *reinterpret_cast<const v8bf*>(p + 16);
  return u.v;
}

// ---------------------------------------------------------------------------
__global__ __launch_bounds__(256) void cfa_cvt_bf16(const float* __restrict__ src,
                                                    bf16* __restrict__ dst, int n) {
  int i = blockIdx.x * blockDim.x + threadIdx.x;
  int stride = gridDim.x * blockDim.x;
  for (; i < n; i += stride) dst[i] = (bf16)src[i];
}

__global__ __launch_bounds__(256) void cfa_zero_f32(float* __restrict__ p, int n) {
  int i = blockIdx.x * blockDim.x + threadIdx.x;
  int stride = gridDim.x * blockDim.x;
  for (; i < n; i += stride) p[i] = 0.0f;
}

// ---------------------------------------------------------------------------
// Phase 1: K/V projection (WMMA) + ||k|| + scatter-add into field.
// grid = (NHEADS, MTOT/64); block = 256 (8 waves).
// wave w: mat = w>>2 (0 -> K, 1 -> V), row-strip = w&3.
__global__ __launch_bounds__(256) void cfa_kv_field(
    const bf16* __restrict__ xb, const bf16* __restrict__ kwb,
    const bf16* __restrict__ vwb, const float* __restrict__ k_bias,
    const float* __restrict__ v_bias, float* __restrict__ field) {
  __shared__ bf16 sA[2][64 * 32];
  __shared__ bf16 sK[2][64 * 32];
  __shared__ bf16 sV[2][64 * 32];
  __shared__ float smag[64];

  const int tid  = threadIdx.x;
  const int lane = tid & 31;
  const int w    = tid >> 5;
  const int mat  = w >> 2;
  const int rs   = w & 3;
  const int h    = blockIdx.x;
  const int rt   = blockIdx.y;
  const int row0 = rt * 64 + rs * 16;
  const int n0   = h * HD;

  // Staging role of this thread: one 16B chunk per tile per stage.
  const int srow = tid >> 2;
  const int sc8  = (tid & 3) * 8;
  const bf16* gA = xb  + (size_t)(rt * 64 + srow) * D_EMB + sc8;
  const bf16* gK = kwb + (size_t)(n0 + srow)      * D_EMB + sc8;
  const bf16* gV = vwb + (size_t)(n0 + srow)      * D_EMB + sc8;
  const int soff = srow * 32 + sc8;

  // Prologue: stage k-step 0 into buffer 0.
  copy16_g2l(gA, &sA[0][soff]);
  copy16_g2l(gK, &sK[0][soff]);
  copy16_g2l(gV, &sV[0][soff]);

  v8f acc0 = {}, acc1 = {}, acc2 = {}, acc3 = {};
  for (int s = 0; s < KSTEPS; ++s) {
    const int cur = s & 1;
    wait_g2l();
    __syncthreads();                       // buffer `cur` fully staged
    if (s + 1 < KSTEPS) {                  // prefetch next stage
      const int ks = (s + 1) * 32;
      copy16_g2l(gA + ks, &sA[cur ^ 1][soff]);
      copy16_g2l(gK + ks, &sK[cur ^ 1][soff]);
      copy16_g2l(gV + ks, &sV[cur ^ 1][soff]);
    }
    const v16bf a  = lds_frag_a(sA[cur], rs, lane);
    const bf16* sW = mat ? sV[cur] : sK[cur];
    acc0 = wmma_bf16(a, lds_frag_b(sW, 0,  lane), acc0);
    acc1 = wmma_bf16(a, lds_frag_b(sW, 16, lane), acc1);
    acc2 = wmma_bf16(a, lds_frag_b(sW, 32, lane), acc2);
    acc3 = wmma_bf16(a, lds_frag_b(sW, 48, lane), acc3);
    __syncthreads();                       // readers done before overwrite
  }

  const int hi = lane >> 4;
  const int cl = lane & 15;

  const float* bias = mat ? v_bias : k_bias;
  const float bb0 = bias[n0 + cl];
  const float bb1 = bias[n0 + 16 + cl];
  const float bb2 = bias[n0 + 32 + cl];
  const float bb3 = bias[n0 + 48 + cl];
#pragma unroll
  for (int j = 0; j < 8; ++j) {
    acc0[j] += bb0; acc1[j] += bb1; acc2[j] += bb2; acc3[j] += bb3;
  }

  if (mat == 0) {
    // ||k|| per row: C-layout row = j + 8*hi, this lane covers 4 cols.
#pragma unroll
    for (int j = 0; j < 8; ++j) {
      float p = acc0[j]*acc0[j] + acc1[j]*acc1[j] + acc2[j]*acc2[j] + acc3[j]*acc3[j];
      p += __shfl_xor(p, 1, 16);
      p += __shfl_xor(p, 2, 16);
      p += __shfl_xor(p, 4, 16);
      p += __shfl_xor(p, 8, 16);
      if (cl == 0) smag[rs * 16 + j + 8 * hi] = sqrtf(p);
    }
  }
  __syncthreads();
  if (mat == 1) {
#pragma unroll
    for (int j = 0; j < 8; ++j) {
      const int rloc = j + 8 * hi;
      const float mg = smag[rs * 16 + rloc];
      const int m = row0 + rloc;
      const int b = m >> 12;              // / NSEQ
      const int n = m & (NSEQ - 1);
      float pos = ((float)n / 4095.0f) * 511.0f;
      int g = (int)pos;
      if (g < 0) g = 0;
      if (g > GFIELD - 1) g = GFIELD - 1;
      float* fb = field + (((size_t)(b * NHEADS + h)) * GFIELD + g) * HD;
      atomicAdd(fb + cl,      acc0[j] * mg);
      atomicAdd(fb + 16 + cl, acc1[j] * mg);
      atomicAdd(fb + 32 + cl, acc2[j] * mg);
      atomicAdd(fb + 48 + cl, acc3[j] * mg);
    }
  }
}

// ---------------------------------------------------------------------------
// Phase 2: causal circular convolution along G (256 nonzero taps, geometric
// weights w(jp) = exp(-(256-jp)/3)/Z). Writes bf16 in (B, G, D) layout.
__global__ __launch_bounds__(256) void cfa_field_conv(
    const float* __restrict__ field, bf16* __restrict__ convb) {
  const int i  = blockIdx.x * blockDim.x + threadIdx.x;  // < B*H*G*HD = 2M
  const int d  = i & (HD - 1);
  const int g  = (i >> 6) & (GFIELD - 1);
  const int bh = i >> 15;
  const int b  = bh >> 4, h = bh & (NHEADS - 1);

  const float r = __expf(-1.0f / 3.0f);
  const float Z = r * (1.0f - __expf(-256.0f / 3.0f)) / (1.0f - r) + 1e-8f;

  const float* fbh = field + (size_t)bh * GFIELD * HD;
  float wgt = r / Z;  // weight at jp = 255
  float acc = 0.0f;
  for (int jp = 255; jp >= 0; --jp) {
    const int src = (g - jp) & (GFIELD - 1);
    acc += wgt * fbh[src * HD + d];
    wgt *= r;
  }
  convb[((size_t)(b * GFIELD + g)) * D_EMB + h * HD + d] = (bf16)acc;
}

// ---------------------------------------------------------------------------
// Phase 3: out = gather(conv)[m,:] @ out_w.T + out_b (WMMA). The gather is
// folded into the per-lane source addresses of the async A-tile staging.
// grid = (D_EMB/64, MTOT/64); block = 256 (8 waves = 4 row-strips x 2 col-halves).
__global__ __launch_bounds__(256) void cfa_out_gemm(
    const bf16* __restrict__ convb, const bf16* __restrict__ owb,
    const float* __restrict__ o_bias, float* __restrict__ out) {
  __shared__ bf16 sA[2][64 * 32];
  __shared__ bf16 sB[2][64 * 32];

  const int tid  = threadIdx.x;
  const int lane = tid & 31;
  const int w    = tid >> 5;
  const int ch   = w >> 2;
  const int rs   = w & 3;
  const int row0 = blockIdx.y * 64 + rs * 16;
  const int n0   = blockIdx.x * 64 + ch * 32;

  // Staging role: gathered A row for m = blockTile + (tid>>2).
  const int srow = tid >> 2;
  const int sc8  = (tid & 3) * 8;
  const int sm   = blockIdx.y * 64 + srow;
  const int sb   = sm >> 12;
  const int sn   = sm & (NSEQ - 1);
  float spos = ((float)sn / 4095.0f) * 511.0f;
  int sg = (int)spos;
  if (sg < 0) sg = 0;
  if (sg > GFIELD - 1) sg = GFIELD - 1;
  const bf16* gA = convb + ((size_t)(sb * GFIELD + sg)) * D_EMB + sc8;
  const bf16* gB = owb + (size_t)(blockIdx.x * 64 + srow) * D_EMB + sc8;
  const int soff = srow * 32 + sc8;

  copy16_g2l(gA, &sA[0][soff]);
  copy16_g2l(gB, &sB[0][soff]);

  v8f acc0 = {}, acc1 = {};
  for (int s = 0; s < KSTEPS; ++s) {
    const int cur = s & 1;
    wait_g2l();
    __syncthreads();
    if (s + 1 < KSTEPS) {
      const int ks = (s + 1) * 32;
      copy16_g2l(gA + ks, &sA[cur ^ 1][soff]);
      copy16_g2l(gB + ks, &sB[cur ^ 1][soff]);
    }
    const v16bf a = lds_frag_a(sA[cur], rs, lane);
    acc0 = wmma_bf16(a, lds_frag_b(sB[cur], ch * 32,      lane), acc0);
    acc1 = wmma_bf16(a, lds_frag_b(sB[cur], ch * 32 + 16, lane), acc1);
    __syncthreads();
  }

  const int hi = lane >> 4;
  const int cl = lane & 15;
  const float bb0 = o_bias[n0 + cl];
  const float bb1 = o_bias[n0 + 16 + cl];
#pragma unroll
  for (int j = 0; j < 8; ++j) {
    const int row = row0 + j + 8 * hi;
    out[(size_t)row * D_EMB + n0 + cl]      = acc0[j] + bb0;
    out[(size_t)row * D_EMB + n0 + 16 + cl] = acc1[j] + bb1;
  }
}

// ---------------------------------------------------------------------------
extern "C" void kernel_launch(void* const* d_in, const int* in_sizes, int n_in,
                              void* d_out, int out_size, void* d_ws, size_t ws_size,
                              hipStream_t stream) {
  (void)in_sizes; (void)n_in; (void)out_size; (void)ws_size;
  const float* x   = (const float*)d_in[0];
  // d_in[1]/d_in[2] = q_w/q_b: unused by the reference output -> skipped.
  const float* k_w = (const float*)d_in[3];
  const float* k_b = (const float*)d_in[4];
  const float* v_w = (const float*)d_in[5];
  const float* v_b = (const float*)d_in[6];
  const float* o_w = (const float*)d_in[7];
  const float* o_b = (const float*)d_in[8];

  char* ws = (char*)d_ws;
  bf16*  xb    = (bf16*)(ws);
  bf16*  kwb   = (bf16*)(ws + (size_t)(32u << 20));
  bf16*  vwb   = (bf16*)(ws + (size_t)(34u << 20));
  bf16*  owb   = (bf16*)(ws + (size_t)(36u << 20));
  float* field = (float*)(ws + (size_t)(40u << 20));
  bf16*  convb = (bf16*)(ws + (size_t)(48u << 20));

  const int NX = MTOT * D_EMB;                  // 16,777,216
  const int NW = D_EMB * D_EMB;                 // 1,048,576
  const int NF = BATCH * NHEADS * GFIELD * HD;  // 2,097,152

  cfa_cvt_bf16<<<4096, 256, 0, stream>>>(x,   xb,  NX);
  cfa_cvt_bf16<<<1024, 256, 0, stream>>>(k_w, kwb, NW);
  cfa_cvt_bf16<<<1024, 256, 0, stream>>>(v_w, vwb, NW);
  cfa_cvt_bf16<<<1024, 256, 0, stream>>>(o_w, owb, NW);
  cfa_zero_f32<<<2048, 256, 0, stream>>>(field, NF);

  cfa_kv_field<<<dim3(NHEADS, MTOT / 64), 256, 0, stream>>>(
      xb, kwb, vwb, k_b, v_b, field);

  cfa_field_conv<<<NF / 256, 256, 0, stream>>>(field, convb);

  cfa_out_gemm<<<dim3(D_EMB / 64, MTOT / 64), 256, 0, stream>>>(
      convb, owb, o_b, (float*)d_out);
}
